// SpatialGNN_28415503630975
// MI455X (gfx1250) — matrix operand order repaired and checked by reading, not compiled, not run on hardware
//
#include <hip/hip_runtime.h>

typedef __attribute__((ext_vector_type(16))) __bf16 v16bf;
typedef __attribute__((ext_vector_type(8)))  float  v8f;

#define N_NODES 50000   // multiple of 16 -> no partial row tiles
#define N_EDGES 800000
#define D_IN    128
#define H_DIM   256
#define B_GRAPHS 64

__device__ __forceinline__ float reluf(float x) { return fmaxf(x, 0.f); }

// ---------------- edge message + scatter add (memory-bound hot loop) --------
template<int DIM>
__global__ void edge_aggr_kernel(const float* __restrict__ h,
                                 const int* __restrict__ src,
                                 const int* __restrict__ dst,
                                 const float* __restrict__ ea,
                                 const float* __restrict__ We,
                                 const float* __restrict__ be,
                                 float* __restrict__ aggr,
                                 int nEdges)
{
    const int lane = threadIdx.x & 31;
    const int wpb  = blockDim.x >> 5;
    int e = blockIdx.x * wpb + (threadIdx.x >> 5);
    const int stride = gridDim.x * wpb;
    for (; e < nEdges; e += stride) {
        const int   s = src[e], t = dst[e];
        const float a = ea[e];
        #pragma unroll
        for (int j = 0; j < DIM / 32; ++j) {
            const int d = lane + 32 * j;                    // coalesced across lanes
            const float m = reluf(h[(long)s * DIM + d] + a * We[d] + be[d]);
            unsafeAtomicAdd(&aggr[(long)t * DIM + d], m);   // global_atomic_add_f32
        }
    }
}

// ---------------- weight transpose + f32->bf16 pre-pass ---------------------
// outT[n*K + k] = (bf16) in[k*Nout + n]   (WT is [Nout][K], rows contiguous)
__global__ void wT_bf16_kernel(const float* __restrict__ in, __bf16* __restrict__ outT,
                               int K, int Nout)
{
    const int idx = blockIdx.x * blockDim.x + threadIdx.x;
    if (idx >= K * Nout) return;
    const int n = idx / K, k = idx - n * K;
    outT[idx] = (__bf16)in[(long)k * Nout + n];
}

// ---------------- bf16 WMMA GEMM: out = [relu]((A (+A2)) @ W + bias) --------
// Block = 8 waves x one 64-wide column block. Each wave computes a 16x64 strip
// (4 accumulators): one A fragment feeds 4 WMMAs. The bf16 B tile (contiguous
// span of pre-transposed WT) is staged via global_load_async_to_lds_b128
// (ASYNCcnt) and read back as aligned 32B v16bf fragments, all 4 loaded as a
// group ahead of the WMMA chain. K/NOUT are compile-time for flat addressing.
template<bool RELU, bool ADD2, int K, int NOUT>
__global__ void gemm_bf16_wmma(const float* __restrict__ A,
                               const float* __restrict__ A2,
                               const __bf16* __restrict__ WT,   // [NOUT][K] bf16
                               const float* __restrict__ bias,
                               float* __restrict__ out,
                               int M)
{
    constexpr int LSTR = 256 + 16;           // padded row: 544B (32B-aligned, conflict-break)
    __shared__ __bf16 Wt[64 * LSTR];         // 64 columns x K

    const int lane = threadIdx.x & 31;
    const int wave = threadIdx.x >> 5;
    constexpr int colBlks = NOUT >> 6;
    const int cb      = blockIdx.x % colBlks;
    const int rowBlk  = blockIdx.x / colBlks;
    const int n0 = cb << 6;

    // ---- async-stage WT rows n0..n0+63 into LDS ----
    constexpr int chunksPerRow = K >> 3;                // 16B chunks per padded row
    constexpr int totalChunks  = 64 * chunksPerRow;     // divisible by 256
    const unsigned ldsBase = (unsigned)(size_t)&Wt[0];  // low 32b of flat = LDS offset
    #pragma unroll 2
    for (int idx = threadIdx.x; idx < totalChunks; idx += 256) {
        const int row = idx / chunksPerRow;
        const int j   = idx - row * chunksPerRow;
        const unsigned ldsAddr = ldsBase + (unsigned)(row * LSTR * 2 + j * 16);
        const unsigned long long gaddr =
            (unsigned long long)(WT + ((size_t)(n0 + row) * K + j * 8));
        asm volatile("global_load_async_to_lds_b128 %0, %1, off"
                     :: "v"(ldsAddr), "v"(gaddr) : "memory");
    }
    asm volatile("s_wait_asynccnt 0x0" ::: "memory");
    __syncthreads();

    const int rowTiles = M >> 4;                        // M % 16 == 0
    const int myTile = rowBlk * 8 + wave;
    if (myTile < rowTiles) {                            // uniform per wave: EXEC all-1s
        const int row0  = myTile << 4;
        const int mA    = row0 + (lane & 15);
        const int kGrpA = (lane >> 4) * 16;             // A: upper lane-group -> K+16
        const int nl    = lane & 15;
        const int kGrpB = (lane >> 4) * 16;             // B: upper lane-group -> K+16

        v8f acc[4] = {};
        #pragma unroll 2
        for (int kt = 0; kt < K; kt += 32) {
            // ---- A fragment: one contiguous 16-float run -> 4x b128 loads ----
            const long baseLo = (long)mA * K + kt + kGrpA;
            float4 q0 = *(const float4*)(A + baseLo);
            float4 q1 = *(const float4*)(A + baseLo + 4);
            float4 q2 = *(const float4*)(A + baseLo + 8);
            float4 q3 = *(const float4*)(A + baseLo + 12);
            if (ADD2) {
                const float4 p0 = *(const float4*)(A2 + baseLo);
                const float4 p1 = *(const float4*)(A2 + baseLo + 4);
                const float4 p2 = *(const float4*)(A2 + baseLo + 8);
                const float4 p3 = *(const float4*)(A2 + baseLo + 12);
                q0.x += p0.x; q0.y += p0.y; q0.z += p0.z; q0.w += p0.w;
                q1.x += p1.x; q1.y += p1.y; q1.z += p1.z; q1.w += p1.w;
                q2.x += p2.x; q2.y += p2.y; q2.z += p2.z; q2.w += p2.w;
                q3.x += p3.x; q3.y += p3.y; q3.z += p3.z; q3.w += p3.w;
            }
            v16bf a;
            a[0]  = (__bf16)q0.x; a[1]  = (__bf16)q0.y; a[2]  = (__bf16)q0.z; a[3]  = (__bf16)q0.w;
            a[4]  = (__bf16)q1.x; a[5]  = (__bf16)q1.y; a[6]  = (__bf16)q1.z; a[7]  = (__bf16)q1.w;
            a[8]  = (__bf16)q2.x; a[9]  = (__bf16)q2.y; a[10] = (__bf16)q2.z; a[11] = (__bf16)q2.w;
            a[12] = (__bf16)q3.x; a[13] = (__bf16)q3.y; a[14] = (__bf16)q3.z; a[15] = (__bf16)q3.w;

            // ---- load all 4 B fragments first, then chain 4 WMMAs ----
            const int kb = kt + kGrpB;
            v16bf bf[4];
            #pragma unroll
            for (int c = 0; c < 4; ++c)
                bf[c] = *(const v16bf*)&Wt[(c * 16 + nl) * LSTR + kb];
            #pragma unroll
            for (int c = 0; c < 4; ++c)
                acc[c] = __builtin_amdgcn_wmma_f32_16x16x32_bf16(false, a, false, bf[c],
                                                                 (short)0, acc[c], false, false);
        }

        #pragma unroll
        for (int c = 0; c < 4; ++c) {
            const int nCol = n0 + c * 16 + nl;
            const float bv = bias[nCol];
            #pragma unroll
            for (int i = 0; i < 8; ++i) {
                const int r = row0 + i + (lane >> 4) * 8;
                float v = acc[c][i] + bv;
                if (RELU) v = reluf(v);
                out[(long)r * NOUT + nCol] = v;
            }
        }
    }
}

// ---------------- LayerNorm(+relu) over 256-wide rows -----------------------
__global__ void ln_relu_kernel(const float* __restrict__ z,
                               const float* __restrict__ g,
                               const float* __restrict__ bt,
                               float* __restrict__ out, int M)
{
    __shared__ float red[H_DIM];
    const int row = blockIdx.x;
    const int tid = threadIdx.x;
    const float v = z[(long)row * H_DIM + tid];
    red[tid] = v; __syncthreads();
    for (int o = H_DIM / 2; o > 0; o >>= 1) { if (tid < o) red[tid] += red[tid + o]; __syncthreads(); }
    const float mu = red[0] * (1.f / H_DIM); __syncthreads();
    const float c = v - mu;
    red[tid] = c * c; __syncthreads();
    for (int o = H_DIM / 2; o > 0; o >>= 1) { if (tid < o) red[tid] += red[tid + o]; __syncthreads(); }
    const float var = red[0] * (1.f / H_DIM);
    out[(long)row * H_DIM + tid] = reluf(c * rsqrtf(var + 1e-5f) * g[tid] + bt[tid]);
}

// ---------------- node head second layer: [N,128] . Wh2 + bh2 ---------------
__global__ void node_head2_kernel(const float* __restrict__ t,
                                  const float* __restrict__ Wh2,
                                  const float* __restrict__ bh2,
                                  float* __restrict__ out, int M)
{
    const int lane = threadIdx.x & 31;
    const int row = blockIdx.x * (blockDim.x >> 5) + (threadIdx.x >> 5);
    if (row >= M) return;
    float p = 0.f;
    #pragma unroll
    for (int j = 0; j < 4; ++j) {
        const int d = lane + 32 * j;
        p += t[(long)row * 128 + d] * Wh2[d];
    }
    for (int o = 16; o > 0; o >>= 1) p += __shfl_xor(p, o, 32);
    if (lane == 0) out[row] = p + bh2[0];
}

// ---------------- graph mean-pool accumulate --------------------------------
__global__ void pool_kernel(const float* __restrict__ h, const int* __restrict__ bvec,
                            float* __restrict__ gsum, float* __restrict__ gcnt, int M)
{
    const int row = blockIdx.x;
    const int tid = threadIdx.x;
    if (row >= M) return;
    const int b = bvec[row];
    unsafeAtomicAdd(&gsum[b * H_DIM + tid], h[(long)row * H_DIM + tid]);
    if (tid == 0) unsafeAtomicAdd(&gcnt[b], 1.f);
}

// ---------------- graph head (B=64, tiny scalar kernel) ---------------------
__global__ void graph_head_kernel(const float* __restrict__ gsum, const float* __restrict__ gcnt,
                                  const float* __restrict__ Wh1, const float* __restrict__ bh1,
                                  const float* __restrict__ Wh2, const float* __restrict__ bh2,
                                  float* __restrict__ out)
{
    __shared__ float red[128];
    const int b = blockIdx.x;
    const int j = threadIdx.x;                       // 0..127
    const float inv = 1.f / fmaxf(gcnt[b], 1.f);
    float hdn = bh1[j];
    for (int k = 0; k < H_DIM; ++k) hdn += gsum[b * H_DIM + k] * inv * Wh1[k * 128 + j];
    hdn = reluf(hdn);
    red[j] = hdn * Wh2[j];
    __syncthreads();
    for (int o = 64; o > 0; o >>= 1) { if (j < o) red[j] += red[j + o]; __syncthreads(); }
    if (j == 0) out[b] = red[0] + bh2[0];
}

extern "C" void kernel_launch(void* const* d_in, const int* in_sizes, int n_in,
                              void* d_out, int out_size, void* d_ws, size_t ws_size,
                              hipStream_t stream)
{
    const float* x   = (const float*)d_in[0];
    const int*  eidx = (const int*)d_in[1];
    const float* ea  = (const float*)d_in[2];
    const int*  bvec = (const int*)d_in[3];
    const float* We0 = (const float*)d_in[4],  *be0 = (const float*)d_in[5];
    const float* W10 = (const float*)d_in[6],  *b10 = (const float*)d_in[7];
    const float* W20 = (const float*)d_in[8],  *b20 = (const float*)d_in[9];
    const float* g0  = (const float*)d_in[10], *bt0 = (const float*)d_in[11];
    const float* We1 = (const float*)d_in[12], *be1 = (const float*)d_in[13];
    const float* W11 = (const float*)d_in[14], *b11 = (const float*)d_in[15];
    const float* W21 = (const float*)d_in[16], *b21 = (const float*)d_in[17];
    const float* g1  = (const float*)d_in[18], *bt1 = (const float*)d_in[19];
    const float* Wh1 = (const float*)d_in[20], *bh1 = (const float*)d_in[21];
    const float* Wh2 = (const float*)d_in[22], *bh2 = (const float*)d_in[23];
    (void)in_sizes; (void)n_in; (void)out_size; (void)ws_size;

    const int* src = eidx;
    const int* dst = eidx + N_EDGES;

    // workspace: 3x [N,256] f32 ping-pong buffers + pool sums + bf16 weightsT
    float* bufA = (float*)d_ws;
    float* bufB = bufA + (size_t)N_NODES * H_DIM;
    float* bufC = bufB + (size_t)N_NODES * H_DIM;
    float* gsum = bufC + (size_t)N_NODES * H_DIM;
    float* gcnt = gsum + (size_t)B_GRAPHS * H_DIM;
    __bf16* W10T = (__bf16*)(gcnt + B_GRAPHS);       // [256][128]
    __bf16* W20T = W10T + 256 * 128;                 // [256][256]
    __bf16* W11T = W20T + 256 * 256;                 // [256][256]
    __bf16* W21T = W11T + 256 * 256;                 // [256][256]
    __bf16* Wh1T = W21T + 256 * 256;                 // [128][256]
    float* outv = (float*)d_out;

    // ---- weight transpose/convert pre-pass (tiny, L2-resident) ----
    wT_bf16_kernel<<<(256 * 128 + 255) / 256, 256, 0, stream>>>(W10, W10T, D_IN,  H_DIM);
    wT_bf16_kernel<<<(256 * 256 + 255) / 256, 256, 0, stream>>>(W20, W20T, H_DIM, H_DIM);
    wT_bf16_kernel<<<(256 * 256 + 255) / 256, 256, 0, stream>>>(W11, W11T, H_DIM, H_DIM);
    wT_bf16_kernel<<<(256 * 256 + 255) / 256, 256, 0, stream>>>(W21, W21T, H_DIM, H_DIM);
    wT_bf16_kernel<<<(256 * 128 + 255) / 256, 256, 0, stream>>>(Wh1, Wh1T, H_DIM, 128);

    const int edgeBlocks = (N_EDGES + 7) / 8;            // 8 waves/block
    const int rowTiles   = N_NODES / 16;                 // 3125
    const int rowBlks    = (rowTiles + 7) / 8;           // 8 row tiles per block
    const int gridH      = rowBlks * (H_DIM / 64);       // 64-wide column blocks
    const int gridHd     = rowBlks * (128 / 64);

    // ---- layer 0 ----
    hipMemsetAsync(bufA, 0, (size_t)N_NODES * D_IN * sizeof(float), stream);
    edge_aggr_kernel<D_IN><<<edgeBlocks, 256, 0, stream>>>(x, src, dst, ea, We0, be0, bufA, N_EDGES);
    gemm_bf16_wmma<true,  true,  D_IN,  H_DIM><<<gridH, 256, 0, stream>>>(x,    bufA,    W10T, b10, bufB, N_NODES);
    gemm_bf16_wmma<false, false, H_DIM, H_DIM><<<gridH, 256, 0, stream>>>(bufB, nullptr, W20T, b20, bufC, N_NODES);
    ln_relu_kernel<<<N_NODES, H_DIM, 0, stream>>>(bufC, g0, bt0, bufA, N_NODES);   // h0 -> bufA

    // ---- layer 1 ----
    hipMemsetAsync(bufB, 0, (size_t)N_NODES * H_DIM * sizeof(float), stream);
    edge_aggr_kernel<H_DIM><<<edgeBlocks, 256, 0, stream>>>(bufA, src, dst, ea, We1, be1, bufB, N_EDGES);
    gemm_bf16_wmma<true,  true,  H_DIM, H_DIM><<<gridH, 256, 0, stream>>>(bufA, bufB,    W11T, b11, bufC, N_NODES);
    gemm_bf16_wmma<false, false, H_DIM, H_DIM><<<gridH, 256, 0, stream>>>(bufC, nullptr, W21T, b21, bufB, N_NODES);
    ln_relu_kernel<<<N_NODES, H_DIM, 0, stream>>>(bufB, g1, bt1, bufC, N_NODES);   // h1 -> bufC

    // ---- node head ----
    gemm_bf16_wmma<true, false, H_DIM, 128><<<gridHd, 256, 0, stream>>>(bufC, nullptr, Wh1T, bh1, bufA, N_NODES);
    node_head2_kernel<<<(N_NODES + 7) / 8, 256, 0, stream>>>(bufA, Wh2, bh2, outv, N_NODES);

    // ---- graph pooling + head ----
    hipMemsetAsync(gsum, 0, (size_t)(B_GRAPHS * H_DIM + B_GRAPHS) * sizeof(float), stream);
    pool_kernel<<<N_NODES, H_DIM, 0, stream>>>(bufC, bvec, gsum, gcnt, N_NODES);
    graph_head_kernel<<<B_GRAPHS, 128, 0, stream>>>(gsum, gcnt, Wh1, bh1, Wh2, bh2, outv + N_NODES);
}
